// CrossAttention1D_28638841929985
// MI455X (gfx1250) — matrix operand order repaired and checked by reading, not compiled, not run on hardware
//
#include <hip/hip_runtime.h>

// ---------------------------------------------------------------------------
// CrossAttention1D for MI455X (gfx1250, wave32, WMMA f16->f32).
// Pipeline: weight-transpose(f16) -> GN stats -> GN apply+transpose(f16)
//           -> LN(f16) -> GEMM Q -> GEMM K -> GEMM V(transposed out)
//           -> flash attention (WMMA QK^T + softmax + WMMA PV)
//           -> GEMM O (+bias +residual, f32 transposed into d_out)
// Workspace use: ~78 MB.
// ---------------------------------------------------------------------------

#define DEV __device__ __forceinline__

constexpr int kB    = 8;
constexpr int kCH   = 512;
constexpr int kL    = 4096;
constexpr int kS    = 256;
constexpr int kCOND = 768;
constexpr int kH    = 8;
constexpr int kHD   = 64;
constexpr int kG    = 32;
constexpr float kEPS = 1e-5f;

typedef __attribute__((ext_vector_type(16))) _Float16 v16h;
typedef __attribute__((ext_vector_type(8)))  _Float16 v8h;
typedef __attribute__((ext_vector_type(8)))  float    v8f;

// --------------------------- WMMA fragment helpers -------------------------
// A-fragment (16x32 f16, M x K): lane holds row M = lane&15.
// half 0 (lanes 0-15):  K = kbase+0..7 (v0..3), kbase+16..23 (v4..7)
// half 1 (lanes 16-31): K = kbase+8..15,        kbase+24..31
DEV v16h load_a_frag(const _Float16* base, int stride, int mbase, int kbase, int lane) {
  const _Float16* p = base + (size_t)(mbase + (lane & 15)) * stride
                           + kbase + ((lane >> 4) << 3);
  v8h lo = *(const v8h*)p;
  v8h hi = *(const v8h*)(p + 16);
  return __builtin_shufflevector(lo, hi, 0,1,2,3,4,5,6,7,8,9,10,11,12,13,14,15);
}

// B-fragment (32x16 f16, K x N), from BT stored [N][K] (column-major B):
// lane holds column N = nbase + (lane&15), same K order as the A-fragment:
// K = kbase + (lane>>4)*8 + 0..7, then kbase + 16 + (lane>>4)*8 + 0..7.
DEV v16h load_b_frag(const _Float16* baseT, int stride, int nbase, int kbase, int lane) {
  const _Float16* p = baseT + (size_t)(nbase + (lane & 15)) * stride
                            + kbase + ((lane >> 4) << 3);
  v8h lo = *(const v8h*)p;
  v8h hi = *(const v8h*)(p + 16);
  return __builtin_shufflevector(lo, hi, 0,1,2,3,4,5,6,7,8,9,10,11,12,13,14,15);
}

DEV v8f wmma16(v16h a, v16h b, v8f c) {
  return __builtin_amdgcn_wmma_f32_16x16x32_f16(
      /*neg_a=*/false, a, /*neg_b=*/false, b,
      /*c_mod=*/(short)0, c, /*reuse_a=*/false, /*reuse_b=*/false);
}

// --------------------------- weight transpose to f16 -----------------------
// WT[n][k] = (f16) W[k][n];  grid*256 == K*N exactly.
__global__ void wtrans_kernel(const float* __restrict__ W, _Float16* __restrict__ WT,
                              int K, int N) {
  int tid = blockIdx.x * 256 + threadIdx.x;
  int n = tid % N, k = tid / N;
  WT[(size_t)n * K + k] = (_Float16)W[(size_t)k * N + n];
}

// --------------------------- GroupNorm stats -------------------------------
// Each (b,g) owns a contiguous 16*4096 = 65536-float region.
__global__ __launch_bounds__(256) void gn_stats_kernel(const float* __restrict__ x,
                                                       float* __restrict__ mu,
                                                       float* __restrict__ rsig) {
  __shared__ float s1[8], s2[8];
  const float* p = x + (size_t)blockIdx.x * 65536;
  float a = 0.f, b = 0.f;
  for (int i = threadIdx.x; i < 65536; i += 256) {
    float v = p[i];
    a += v; b += v * v;
  }
  #pragma unroll
  for (int o = 16; o >= 1; o >>= 1) { a += __shfl_xor(a, o); b += __shfl_xor(b, o); }
  int wid = threadIdx.x >> 5, lane = threadIdx.x & 31;
  if (lane == 0) { s1[wid] = a; s2[wid] = b; }
  __syncthreads();
  if (threadIdx.x == 0) {
    float sa = 0.f, sb = 0.f;
    #pragma unroll
    for (int i = 0; i < 8; ++i) { sa += s1[i]; sb += s2[i]; }
    float m = sa / 65536.f;
    float var = sb / 65536.f - m * m;
    mu[blockIdx.x] = m;
    rsig[blockIdx.x] = rsqrtf(var + kEPS);
  }
}

// --------------------------- GroupNorm apply + transpose -------------------
// Reads x[b][c][l] (coalesced along l), writes xq[b][l][c] f16 via LDS tile.
__global__ __launch_bounds__(256) void gn_apply_kernel(const float* __restrict__ x,
                                                       const float* __restrict__ mu,
                                                       const float* __restrict__ rsig,
                                                       const float* __restrict__ gg,
                                                       const float* __restrict__ gb,
                                                       _Float16* __restrict__ xq) {
  __shared__ _Float16 tile[32][520];   // 32 l-rows x 512 ch (+8 pad)
  int b  = blockIdx.x >> 7;
  int l0 = (blockIdx.x & 127) * 32;
  int li = threadIdx.x & 31;
  int c0 = threadIdx.x >> 5;           // 0..7
  for (int c = c0; c < kCH; c += 8) {
    float v = x[((size_t)b * kCH + c) * kL + l0 + li];
    int gi = b * kG + (c >> 4);
    float y = (v - mu[gi]) * rsig[gi] * gg[c] + gb[c];
    tile[li][c] = (_Float16)y;
  }
  __syncthreads();
  int row = threadIdx.x >> 3;          // 0..31
  int seg = threadIdx.x & 7;           // 0..7 -> 64 halfs each
  _Float16* dst = xq + ((size_t)b * kL + l0 + row) * kCH + seg * 64;
  #pragma unroll
  for (int j = 0; j < 8; ++j)
    *(v8h*)(dst + j * 8) = *(const v8h*)&tile[row][seg * 64 + j * 8];
}

// --------------------------- LayerNorm -------------------------------------
__global__ __launch_bounds__(256) void ln_kernel(const float* __restrict__ cond,
                                                 const float* __restrict__ lg,
                                                 const float* __restrict__ lb,
                                                 _Float16* __restrict__ cn) {
  __shared__ float s1[8], s2[8];
  const float* xr = cond + (size_t)blockIdx.x * kCOND;
  float a = 0.f, b = 0.f;
  float v0 = xr[threadIdx.x], v1 = xr[threadIdx.x + 256], v2 = xr[threadIdx.x + 512];
  a = v0 + v1 + v2;
  b = v0 * v0 + v1 * v1 + v2 * v2;
  #pragma unroll
  for (int o = 16; o >= 1; o >>= 1) { a += __shfl_xor(a, o); b += __shfl_xor(b, o); }
  int wid = threadIdx.x >> 5, lane = threadIdx.x & 31;
  if (lane == 0) { s1[wid] = a; s2[wid] = b; }
  __syncthreads();
  float sa = 0.f, sb = 0.f;
  #pragma unroll
  for (int i = 0; i < 8; ++i) { sa += s1[i]; sb += s2[i]; }
  float m  = sa / (float)kCOND;
  float rs = rsqrtf(sb / (float)kCOND - m * m + kEPS);
  _Float16* orow = cn + (size_t)blockIdx.x * kCOND;
  orow[threadIdx.x]       = (_Float16)((v0 - m) * rs * lg[threadIdx.x]       + lb[threadIdx.x]);
  orow[threadIdx.x + 256] = (_Float16)((v1 - m) * rs * lg[threadIdx.x + 256] + lb[threadIdx.x + 256]);
  orow[threadIdx.x + 512] = (_Float16)((v2 - m) * rs * lg[threadIdx.x + 512] + lb[threadIdx.x + 512]);
}

// --------------------------- Generic WMMA GEMM -----------------------------
// C[M,N] = A[M,K](f16,row-major) * BT[N,K]^T + bias.  One wave = 32x64 tile.
// MODE 0: f16 out, row-major [M][N]            (Q, K projections)
// MODE 1: f16 out, transposed [m/LD][N][m%LD]  (V stored as [B][C][S])
// MODE 2: f32 out, transposed + residual       (O projection into d_out)
template <int MODE>
__global__ __launch_bounds__(256) void wmma_gemm(const _Float16* __restrict__ A,
                                                 const _Float16* __restrict__ BT,
                                                 const float* __restrict__ bias,
                                                 void* __restrict__ outv,
                                                 const float* __restrict__ resid,
                                                 int M, int N, int K, int LD) {
  const int wid = threadIdx.x >> 5, lane = threadIdx.x & 31;
  const int ntn = N >> 6;
  const int w  = blockIdx.x * 8 + wid;
  const int m0 = (w / ntn) * 32;
  const int n0 = (w % ntn) * 64;

  v8f acc[2][4] = {};
  for (int k = 0; k < K; k += 32) {
    v16h a0 = load_a_frag(A, K, m0,      k, lane);
    v16h a1 = load_a_frag(A, K, m0 + 16, k, lane);
    v16h b0 = load_b_frag(BT, K, n0 +  0, k, lane);
    v16h b1 = load_b_frag(BT, K, n0 + 16, k, lane);
    v16h b2 = load_b_frag(BT, K, n0 + 32, k, lane);
    v16h b3 = load_b_frag(BT, K, n0 + 48, k, lane);
    acc[0][0] = wmma16(a0, b0, acc[0][0]);
    acc[0][1] = wmma16(a0, b1, acc[0][1]);
    acc[0][2] = wmma16(a0, b2, acc[0][2]);
    acc[0][3] = wmma16(a0, b3, acc[0][3]);
    acc[1][0] = wmma16(a1, b0, acc[1][0]);
    acc[1][1] = wmma16(a1, b1, acc[1][1]);
    acc[1][2] = wmma16(a1, b2, acc[1][2]);
    acc[1][3] = wmma16(a1, b3, acc[1][3]);
    if (k + 32 < K) {
      __builtin_prefetch(A  + (size_t)(m0 + (lane & 15)) * K + k + 32, 0, 0);
      __builtin_prefetch(BT + (size_t)(n0 + (lane & 15)) * K + k + 32, 0, 0);
    }
  }

  const int hh = lane >> 4, nl = lane & 15;
  #pragma unroll
  for (int nt = 0; nt < 4; ++nt) {
    float bv = bias[n0 + nt * 16 + nl];
    #pragma unroll
    for (int mt = 0; mt < 2; ++mt)
      #pragma unroll
      for (int r = 0; r < 8; ++r) acc[mt][nt][r] += bv;
  }

  if constexpr (MODE == 0) {
    // LDS transpose for fully coalesced f16 row stores.
    __shared__ _Float16 sT[8][32][80];
    #pragma unroll
    for (int mt = 0; mt < 2; ++mt)
      #pragma unroll
      for (int nt = 0; nt < 4; ++nt)
        #pragma unroll
        for (int r = 0; r < 8; ++r)
          sT[wid][mt * 16 + hh * 8 + r][nt * 16 + nl] = (_Float16)acc[mt][nt][r];
    _Float16* out = (_Float16*)outv;
    _Float16* dst = out + (size_t)(m0 + lane) * N + n0;
    #pragma unroll
    for (int seg = 0; seg < 8; ++seg)
      *(v8h*)(dst + seg * 8) = *(const v8h*)&sT[wid][lane][seg * 8];
  } else if constexpr (MODE == 1) {
    _Float16* out = (_Float16*)outv;
    #pragma unroll
    for (int mt = 0; mt < 2; ++mt)
      #pragma unroll
      for (int nt = 0; nt < 4; ++nt) {
        int mb  = m0 + mt * 16 + hh * 8;           // 8 contiguous m per lane
        int col = n0 + nt * 16 + nl;
        size_t idx = (size_t)(mb / LD) * N * LD + (size_t)col * LD + (mb % LD);
        v8h h8;
        #pragma unroll
        for (int r = 0; r < 8; ++r) h8[r] = (_Float16)acc[mt][nt][r];
        *(v8h*)(out + idx) = h8;
      }
  } else {
    float* out = (float*)outv;
    #pragma unroll
    for (int mt = 0; mt < 2; ++mt)
      #pragma unroll
      for (int nt = 0; nt < 4; ++nt) {
        int mb  = m0 + mt * 16 + hh * 8;
        int col = n0 + nt * 16 + nl;
        size_t idx = (size_t)(mb / LD) * N * LD + (size_t)col * LD + (mb % LD);
        const float* rp = resid + idx;
        float4 x0 = *(const float4*)rp;
        float4 x1 = *(const float4*)(rp + 4);
        x0.x += acc[mt][nt][0]; x0.y += acc[mt][nt][1];
        x0.z += acc[mt][nt][2]; x0.w += acc[mt][nt][3];
        x1.x += acc[mt][nt][4]; x1.y += acc[mt][nt][5];
        x1.z += acc[mt][nt][6]; x1.w += acc[mt][nt][7];
        *(float4*)(out + idx)     = x0;
        *(float4*)(out + idx + 4) = x1;
      }
  }
}

// --------------------------- Flash-style cross attention -------------------
// One wave per 16-query tile per (b,h). S=256 keys scored at once (16 accums),
// f32 softmax in the WMMA D layout, attn -> LDS -> A-fragments for P*V.
__global__ __launch_bounds__(128) void flash_attn_kernel(const _Float16* __restrict__ q,
                                                         const _Float16* __restrict__ kf,
                                                         const _Float16* __restrict__ vT,
                                                         _Float16* __restrict__ ao) {
  __shared__ _Float16 sAttn[4][16][264];   // padded: row stride 528B -> bank-safe
  __shared__ _Float16 sOut[4][16][80];
  const int wid = threadIdx.x >> 5, lane = threadIdx.x & 31;
  const int w   = blockIdx.x * 4 + wid;
  const int bh  = w >> 8;                  // 256 m-tiles per (b,h)
  const int mt  = w & 255;
  const int b   = bh >> 3, h = bh & 7;
  const int l0  = mt * 16;

  const _Float16* qbase = q  + ((size_t)b * kL + l0) * kCH + h * kHD;
  const _Float16* kbase = kf + (size_t)b * kS * kCH + h * kHD;
  const _Float16* vbase = vT + (size_t)b * kCH * kS + (size_t)h * kHD * kS;

  v16h qa0 = load_a_frag(qbase, kCH, 0, 0,  lane);
  v16h qa1 = load_a_frag(qbase, kCH, 0, 32, lane);

  v8f sc[16] = {};
  #pragma unroll
  for (int t = 0; t < 16; ++t) {
    v16h kb0 = load_b_frag(kbase, kCH, t * 16, 0,  lane);
    v16h kb1 = load_b_frag(kbase, kCH, t * 16, 32, lane);
    sc[t] = wmma16(qa0, kb0, sc[t]);
    sc[t] = wmma16(qa1, kb1, sc[t]);
  }

  // softmax over S=256 (scale applied inside exp); row = hh*8 + r in D layout
  const float scale = 0.125f;  // 64^-0.5
  float rinv[8];
  #pragma unroll
  for (int r = 0; r < 8; ++r) {
    float m = -3.4e38f;
    #pragma unroll
    for (int t = 0; t < 16; ++t) m = fmaxf(m, sc[t][r]);
    #pragma unroll
    for (int o = 8; o >= 1; o >>= 1) m = fmaxf(m, __shfl_xor(m, o));  // width-16
    float s = 0.f;
    #pragma unroll
    for (int t = 0; t < 16; ++t) {
      float e = __expf((sc[t][r] - m) * scale);
      sc[t][r] = e;
      s += e;
    }
    #pragma unroll
    for (int o = 8; o >= 1; o >>= 1) s += __shfl_xor(s, o);
    rinv[r] = 1.f / s;
  }

  const int hh = lane >> 4, nl = lane & 15;
  #pragma unroll
  for (int t = 0; t < 16; ++t)
    #pragma unroll
    for (int r = 0; r < 8; ++r)
      sAttn[wid][hh * 8 + r][t * 16 + nl] = (_Float16)(sc[t][r] * rinv[r]);

  // out[m][d] = attn[m][s] * v[s][d]; B-frags contiguous thanks to vT layout
  v8f ov[4] = {};
  #pragma unroll
  for (int ks = 0; ks < 8; ++ks) {
    v16h af = load_a_frag(&sAttn[wid][0][0], 264, 0, ks * 32, lane);
    #pragma unroll
    for (int nt = 0; nt < 4; ++nt) {
      v16h bf = load_b_frag(vbase, kS, nt * 16, ks * 32, lane);
      ov[nt] = wmma16(af, bf, ov[nt]);
    }
  }

  #pragma unroll
  for (int nt = 0; nt < 4; ++nt)
    #pragma unroll
    for (int r = 0; r < 8; ++r)
      sOut[wid][hh * 8 + r][nt * 16 + nl] = (_Float16)ov[nt][r];

  // coalesced store: each lane writes 64B of one output row
  int row = lane >> 1, seg = lane & 1;
  _Float16* dst = ao + ((size_t)b * kL + l0 + row) * kCH + h * kHD + seg * 32;
  const _Float16* src = &sOut[wid][row][seg * 32];
  #pragma unroll
  for (int j = 0; j < 4; ++j)
    *(v8h*)(dst + j * 8) = *(const v8h*)(src + j * 8);
}

// --------------------------- launcher --------------------------------------
extern "C" void kernel_launch(void* const* d_in, const int* in_sizes, int n_in,
                              void* d_out, int out_size, void* d_ws, size_t ws_size,
                              hipStream_t stream) {
  const float* x    = (const float*)d_in[0];
  const float* cond = (const float*)d_in[1];
  const float* gn_g = (const float*)d_in[2];
  const float* gn_b = (const float*)d_in[3];
  const float* ln_g = (const float*)d_in[4];
  const float* ln_b = (const float*)d_in[5];
  const float* Wq   = (const float*)d_in[6];
  const float* bq   = (const float*)d_in[7];
  const float* Wk   = (const float*)d_in[8];
  const float* bk   = (const float*)d_in[9];
  const float* Wv   = (const float*)d_in[10];
  const float* bv   = (const float*)d_in[11];
  const float* Wo   = (const float*)d_in[12];
  const float* bo   = (const float*)d_in[13];

  char* ws = (char*)d_ws;
  size_t off = 0;
  auto carve = [&](size_t bytes) -> char* {
    char* p = ws + off;
    off += (bytes + 255) & ~(size_t)255;
    return p;
  };
  float*    mu  = (float*)   carve(kB * kG * sizeof(float));
  float*    rs  = (float*)   carve(kB * kG * sizeof(float));
  _Float16* WqT = (_Float16*)carve((size_t)kCH  * kCH * 2);
  _Float16* WkT = (_Float16*)carve((size_t)kCOND* kCH * 2);
  _Float16* WvT = (_Float16*)carve((size_t)kCOND* kCH * 2);
  _Float16* WoT = (_Float16*)carve((size_t)kCH  * kCH * 2);
  _Float16* cn  = (_Float16*)carve((size_t)kB * kS * kCOND * 2);
  _Float16* kf  = (_Float16*)carve((size_t)kB * kS * kCH * 2);
  _Float16* vT  = (_Float16*)carve((size_t)kB * kCH * kS * 2);
  _Float16* xq  = (_Float16*)carve((size_t)kB * kL * kCH * 2);
  _Float16* qf  = (_Float16*)carve((size_t)kB * kL * kCH * 2);
  _Float16* ao  = xq;  // xq is dead after the Q GEMM; reuse for attention out

  // 1) f16 transposed weights
  wtrans_kernel<<<(kCH  * kCH) / 256, 256, 0, stream>>>(Wq, WqT, kCH,   kCH);
  wtrans_kernel<<<(kCOND* kCH) / 256, 256, 0, stream>>>(Wk, WkT, kCOND, kCH);
  wtrans_kernel<<<(kCOND* kCH) / 256, 256, 0, stream>>>(Wv, WvT, kCOND, kCH);
  wtrans_kernel<<<(kCH  * kCH) / 256, 256, 0, stream>>>(Wo, WoT, kCH,   kCH);

  // 2) normalizations
  gn_stats_kernel<<<kB * kG, 256, 0, stream>>>(x, mu, rs);
  gn_apply_kernel<<<kB * (kL / 32), 256, 0, stream>>>(x, mu, rs, gn_g, gn_b, xq);
  ln_kernel<<<kB * kS, 256, 0, stream>>>(cond, ln_g, ln_b, cn);

  // 3) projections (wave tile 32x64; 8 waves/block)
  const int Mq = kB * kL;   // 32768
  const int Mc = kB * kS;   // 2048
  wmma_gemm<0><<<(Mq / 32) * (kCH / 64) / 8, 256, 0, stream>>>(
      xq, WqT, bq, qf, nullptr, Mq, kCH, kCH, 0);
  wmma_gemm<0><<<(Mc / 32) * (kCH / 64) / 8, 256, 0, stream>>>(
      cn, WkT, bk, kf, nullptr, Mc, kCH, kCOND, 0);
  wmma_gemm<1><<<(Mc / 32) * (kCH / 64) / 8, 256, 0, stream>>>(
      cn, WvT, bv, vT, nullptr, Mc, kCH, kCOND, kS);

  // 4) attention: B*H*(L/16) waves, 4 waves per block
  flash_attn_kernel<<<(kB * kH * (kL / 16)) / 4, 128, 0, stream>>>(qf, kf, vT, ao);

  // 5) output projection + bias + residual, transposed f32 into d_out
  wmma_gemm<2><<<(Mq / 32) * (kCH / 64) / 8, 256, 0, stream>>>(
      ao, WoT, bo, d_out, x, Mq, kCH, kCH, kL);
}
